// Transformer_11106785428017
// MI455X (gfx1250) — matrix operand order repaired and checked by reading, not compiled, not run on hardware
//
#include <hip/hip_runtime.h>
#include <math.h>

#define L_  6
#define E_  256
#define H_  8
#define DH_ 32
#define F_  2048
#define NQ_ 300
#define B_  8
#define M_  1024
#define NTQ (NQ_*B_)   /* 2400 query tokens  */
#define NTM (M_*B_)    /* 8192 memory tokens */

typedef __bf16 bf16;
typedef __attribute__((ext_vector_type(16))) __bf16 bf16x16;
typedef __attribute__((ext_vector_type(8)))  __bf16 bf16x8;
typedef __attribute__((ext_vector_type(8)))  float  f32x8;

__device__ __forceinline__ bf16 f2bf(float x){ return (bf16)x; }
__device__ __forceinline__ int  imin(int a, int b){ return a < b ? a : b; }

// ---------------------------------------------------------------------------
// DPP16 cross-lane reductions within a 16-lane half-row (no LDS, no waits).
// quad_perm(1,0,3,2)=0xB1 (xor1), quad_perm(2,3,0,1)=0x4E (xor2),
// row_half_mirror=0x141, row_mirror=0x140.
// ---------------------------------------------------------------------------
template<int CTRL>
__device__ __forceinline__ float dpp16(float x){
  return __builtin_bit_cast(float,
      __builtin_amdgcn_update_dpp(0, __builtin_bit_cast(int, x),
                                  CTRL, 0xF, 0xF, true));
}
__device__ __forceinline__ float redmax16(float x){
  x = fmaxf(x, dpp16<0xB1>(x));
  x = fmaxf(x, dpp16<0x4E>(x));
  x = fmaxf(x, dpp16<0x141>(x));
  x = fmaxf(x, dpp16<0x140>(x));
  return x;
}
__device__ __forceinline__ float redsum16(float x){
  x += dpp16<0xB1>(x);
  x += dpp16<0x4E>(x);
  x += dpp16<0x141>(x);
  x += dpp16<0x140>(x);
  return x;
}

// ---------------------------------------------------------------------------
// WMMA fragment loaders (16x16x32 bf16, wave32; this file's contiguous k order, not AMD's table order)
// A (16xK=32): lane 0-15 row M=lane, elems = K 0..15
//              lane16-31 row M=lane-16, elems = K 16..31
// B (K=32x16): lane 0-15 col N=lane, elems = K 0..15; lane16-31 K 16..31
// ---------------------------------------------------------------------------
__device__ __forceinline__ bf16x16 frag_a_lds(const bf16* p, int ld, int lane){
  int r  = lane & 15;
  int h8 = (lane >> 4) << 4;
  const bf16* q = p + r*ld + h8;
  bf16x8 lo = *(const bf16x8*)(q);
  bf16x8 hi = *(const bf16x8*)(q + 8);
  bf16x16 f;
#pragma unroll
  for (int i = 0; i < 8; ++i){ f[i] = lo[i]; f[i+8] = hi[i]; }
  return f;
}

__device__ __forceinline__ bf16x16 frag_b_lds(const bf16* p, int ld, int lane){
  int c   = lane & 15;
  int h16 = (lane >> 4) << 4;
  const bf16* q = p + c*ld + h16;
  bf16x8 lo = *(const bf16x8*)(q);
  bf16x8 hi = *(const bf16x8*)(q + 8);
  bf16x16 f;
#pragma unroll
  for (int i = 0; i < 8; ++i){ f[i] = lo[i]; f[i+8] = hi[i]; }
  return f;
}

__device__ __forceinline__ bf16x8 pack8(float4 a, float4 b){
  bf16x8 h;
  h[0]=f2bf(a.x); h[1]=f2bf(a.y); h[2]=f2bf(a.z); h[3]=f2bf(a.w);
  h[4]=f2bf(b.x); h[5]=f2bf(b.y); h[6]=f2bf(b.z); h[7]=f2bf(b.w);
  return h;
}

// B-fragment from an fp32 LDS tile (row-major [c][k]), converting on the fly.
__device__ __forceinline__ bf16x16 frag_b_f32lds(const float* p, int ld, int lane){
  int c   = lane & 15;
  int h16 = (lane >> 4) << 4;
  const float* q = p + c*ld + h16;
  float4 x0 = *(const float4*)(q);
  float4 x1 = *(const float4*)(q + 4);
  float4 x2 = *(const float4*)(q + 8);
  float4 x3 = *(const float4*)(q + 12);
  bf16x8 lo = pack8(x0, x1), hi = pack8(x2, x3);
  bf16x16 f;
#pragma unroll
  for (int i = 0; i < 8; ++i){ f[i] = lo[i]; f[i+8] = hi[i]; }
  return f;
}

// ---------------------------------------------------------------------------
// GEMM: Y[N,Mo] = act( alpha * (X[N,K] @ W[Mo,K]^T + bias[Mo]) )
// 128x64 block tile, BK=32, 8 waves; each wave: one 16-row A-frag x 4 B-frags
// VECK=1: K % 32 == 0 -> branch-free float4 staging (rows clamped; OOB rows
// only ever feed OOB outputs, which are never stored).
// ---------------------------------------------------------------------------
#define BK  32
#define LDT 40   // padded LDS stride in halves (80B, 16B aligned)

template<int VECK>
__global__ __launch_bounds__(256)
void gemm_bias(const float* __restrict__ X, const float* __restrict__ W,
               const float* __restrict__ bias, float* __restrict__ Y,
               int N, int K, int Mo, float alpha, int relu)
{
  __shared__ bf16 sX[128*LDT];
  __shared__ bf16 sW[64*LDT];
  const int n0   = blockIdx.x * 128;
  const int m0   = blockIdx.y * 64;
  const int tid  = threadIdx.x;
  const int lane = tid & 31;
  const int wave = tid >> 5;            // M tile = wave*16

  const f32x8 fzero = {0.f,0.f,0.f,0.f,0.f,0.f,0.f,0.f};
  f32x8 acc[4];
#pragma unroll
  for (int i = 0; i < 4; ++i) acc[i] = fzero;

  // staging assignment: X 128x32 (2 thr/row, 16 cols each), W 64x32 (4 thr/row)
  const int xr = tid >> 1, xc = (tid & 1) << 4;
  const int wr = tid >> 2, wc = (tid & 3) << 3;
  const float* xp = X + (size_t)imin(n0 + xr, N - 1) * K;
  const float* wp = W + (size_t)imin(m0 + wr, Mo - 1) * K;

  for (int k0 = 0; k0 < K; k0 += BK){
    if (VECK){
      float4 a0 = *(const float4*)(xp + k0 + xc);
      float4 a1 = *(const float4*)(xp + k0 + xc + 4);
      float4 a2 = *(const float4*)(xp + k0 + xc + 8);
      float4 a3 = *(const float4*)(xp + k0 + xc + 12);
      *(bf16x8*)(&sX[xr*LDT + xc])     = pack8(a0, a1);
      *(bf16x8*)(&sX[xr*LDT + xc + 8]) = pack8(a2, a3);
      float4 b0 = *(const float4*)(wp + k0 + wc);
      float4 b1 = *(const float4*)(wp + k0 + wc + 4);
      *(bf16x8*)(&sW[wr*LDT + wc]) = pack8(b0, b1);
      if (k0 + BK < K){
        __builtin_prefetch(xp + k0 + BK + xc, 0, 3);
        __builtin_prefetch(wp + k0 + BK + wc, 0, 3);
      }
    } else {
      // clamped addresses + VALU select (no branches); needed only for K=4
#pragma unroll
      for (int i = 0; i < 16; ++i){
        const int gk = k0 + xc + i;
        const float v = xp[imin(gk, K - 1)];
        sX[xr*LDT + xc + i] = f2bf(gk < K ? v : 0.f);
      }
#pragma unroll
      for (int i = 0; i < 8; ++i){
        const int gk = k0 + wc + i;
        const float v = wp[imin(gk, K - 1)];
        sW[wr*LDT + wc + i] = f2bf(gk < K ? v : 0.f);
      }
    }
    __syncthreads();

    bf16x16 a = frag_a_lds(&sX[(wave*16)*LDT], LDT, lane);
#pragma unroll
    for (int nn = 0; nn < 4; ++nn){
      bf16x16 bfr = frag_b_lds(&sW[(nn*16)*LDT], LDT, lane);
      acc[nn] = __builtin_amdgcn_wmma_f32_16x16x32_bf16(false, a, false, bfr,
                                                        (short)0, acc[nn], false, false);
    }
    __syncthreads();
  }

  // epilogue: C/D layout, row = v + 8*(lane>=16), col = lane&15
  const int rowadd = (lane >> 4) << 3;
  const int col    = lane & 15;
#pragma unroll
  for (int nn = 0; nn < 4; ++nn){
    const int gc = m0 + nn*16 + col;
    if (gc >= Mo) continue;
    const float bv = bias[gc];
#pragma unroll
    for (int v = 0; v < 8; ++v){
      const int gr = n0 + wave*16 + rowadd + v;
      if (gr < N){
        float r = alpha * (acc[nn][v] + bv);
        if (relu) r = fmaxf(r, 0.f);
        Y[(size_t)gr*Mo + gc] = r;
      }
    }
  }
}

// ---------------------------------------------------------------------------
// Flash attention: one (b,h) per block, 8 waves x 16-query tiles, 32-key
// chunks. Dh=32 == WMMA K. K/V chunks (4KB each, fp32) are staged via
// GLOBAL_LOAD_ASYNC_TO_LDS_B128 (one b128 per thread per matrix, ASYNCcnt-
// tracked, no VGPR round trip), shared by all 8 waves; fragments convert
// fp32->bf16 on the fly. Softmax stats via DPP16. OOB keys neutralized by the
// -1e30 score mask (also zeroes their P coefficients for P.V).
// ---------------------------------------------------------------------------
__global__ __launch_bounds__(256)
void attn_kernel(const float* __restrict__ Qp, const float* __restrict__ Kp,
                 const float* __restrict__ Vp, float* __restrict__ O,
                 int Sq, int Sk)
{
  __shared__ float sKf[32*36];     // [key][dim] fp32 (stride 36: 144B, 16B-aligned)
  __shared__ float sVf[32*36];     // [key][dim] fp32
  __shared__ bf16  sP [8][16*40];  // per-wave P staging (bf16)
  const int bh   = blockIdx.x;
  const int b    = bh & (B_ - 1);
  const int h    = bh >> 3;
  const int tid  = threadIdx.x;
  const int wave = tid >> 5;
  const int lane = tid & 31;
  const int q0   = (blockIdx.y * 8 + wave) * 16;
  const int r    = lane & 15;
  const int h8   = (lane >> 4) << 3;
  const int h16  = (lane >> 4) << 4;

  // async staging assignment: 32 keys x 32 dims, one b128 (4 floats) / thread
  const int skey = tid >> 3;          // 0..31
  const int sdc  = (tid & 7) << 2;    // 0,4,...,28
  const unsigned ldsK = (unsigned)(size_t)(void*)&sKf[skey*36 + sdc];
  const unsigned ldsV = (unsigned)(size_t)(void*)&sVf[skey*36 + sdc];

  // Q fragment (A layout), vectorized from global, rows clamped
  bf16x16 qf;
  {
    const float* qb = Qp + ((size_t)imin(q0 + r, Sq - 1)*B_ + b)*E_ + h*DH_;
    float4 a0 = *(const float4*)(qb + h16);
    float4 a1 = *(const float4*)(qb + h16 + 4);
    float4 a2 = *(const float4*)(qb + h16 + 8);
    float4 a3 = *(const float4*)(qb + h16 + 12);
    bf16x8 lo = pack8(a0, a1), hi = pack8(a2, a3);
#pragma unroll
    for (int i = 0; i < 8; ++i){ qf[i] = lo[i]; qf[i+8] = hi[i]; }
  }

  f32x8 o0 = {0.f,0.f,0.f,0.f,0.f,0.f,0.f,0.f};
  f32x8 o1 = o0;
  float mrow[8], lrow[8];
#pragma unroll
  for (int v = 0; v < 8; ++v){ mrow[v] = -1e30f; lrow[v] = 0.f; }
  bf16* pw = &sP[wave][0];

  for (int kc = 0; kc < Sk; kc += 32){
    // ---- async global->LDS staging of K and V chunk (rows clamped) ----
    {
      const size_t krow = ((size_t)imin(kc + skey, Sk - 1)*B_ + b)*E_ + h*DH_ + sdc;
      asm volatile("global_load_async_to_lds_b128 %0, %1, off"
                   :: "v"(ldsK),
                      "v"((unsigned long long)(size_t)(Kp + krow))
                   : "memory");
      asm volatile("global_load_async_to_lds_b128 %0, %1, off"
                   :: "v"(ldsV),
                      "v"((unsigned long long)(size_t)(Vp + krow))
                   : "memory");
      asm volatile("s_wait_asynccnt 0x0" ::: "memory");
    }
    __syncthreads();

    bf16x16 kf0 = frag_b_f32lds(sKf,         36, lane);  // keys kc..kc+15
    bf16x16 kf1 = frag_b_f32lds(sKf + 16*36, 36, lane);  // keys kc+16..kc+31

    const f32x8 z = {0.f,0.f,0.f,0.f,0.f,0.f,0.f,0.f};
    f32x8 s0 = __builtin_amdgcn_wmma_f32_16x16x32_bf16(false, qf, false, kf0,
                                                       (short)0, z, false, false);
    f32x8 s1 = __builtin_amdgcn_wmma_f32_16x16x32_bf16(false, qf, false, kf1,
                                                       (short)0, z, false, false);
    // online softmax; rows across (vgpr v, half-wave), cols across 16 lanes
#pragma unroll
    for (int v = 0; v < 8; ++v){
      float a0 = (kc + r      < Sk) ? s0[v] : -1e30f;
      float a1 = (kc + 16 + r < Sk) ? s1[v] : -1e30f;
      const float cm = redmax16(fmaxf(a0, a1));
      const float nm = fmaxf(mrow[v], cm);
      const float sf = __expf(mrow[v] - nm);
      const float p0 = __expf(a0 - nm);
      const float p1 = __expf(a1 - nm);
      mrow[v] = nm;
      lrow[v] = lrow[v]*sf + p0 + p1;
      o0[v] *= sf;  o1[v] *= sf;
      const int prow = v + h8;
      pw[prow*40 + r]      = f2bf(p0);
      pw[prow*40 + 16 + r] = f2bf(p1);
    }
    // P as A-frag (per-wave LDS re-layout); V as B-frags gathered from fp32 LDS
    bf16x16 pf = frag_a_lds(pw, 40, lane);
    bf16x16 vf0, vf1;
#pragma unroll
    for (int i = 0; i < 16; ++i){
      vf0[i] = f2bf(sVf[(h16 + i)*36 + r]);        // dims 0..15
      vf1[i] = f2bf(sVf[(h16 + i)*36 + 16 + r]);   // dims 16..31
    }
    o0 = __builtin_amdgcn_wmma_f32_16x16x32_bf16(false, pf, false, vf0,
                                                 (short)0, o0, false, false);
    o1 = __builtin_amdgcn_wmma_f32_16x16x32_bf16(false, pf, false, vf1,
                                                 (short)0, o1, false, false);
    __syncthreads();   // protect sKf/sVf against next chunk's staging
  }

#pragma unroll
  for (int v = 0; v < 8; ++v){
    const float inv = 1.f / redsum16(lrow[v]);
    const int qrow = q0 + v + h8;
    if (qrow < Sq){
      float* ob = O + ((size_t)qrow*B_ + b)*E_ + h*DH_;
      ob[r]      = o0[v] * inv;
      ob[16 + r] = o1[v] * inv;
    }
  }
}

// ---------------------------------------------------------------------------
// Elementwise helpers
// ---------------------------------------------------------------------------
__global__ void add2_kernel(const float* __restrict__ a, const float* __restrict__ b,
                            float* __restrict__ y, int n)
{
  int i = blockIdx.x*256 + threadIdx.x;
  if (i < n) y[i] = a[i] + b[i];
}

// Y[row] = LayerNorm(A[row] + B[row]) * w + bias, E_=256 features, 1 block/row
__global__ __launch_bounds__(256)
void add_ln_kernel(const float* __restrict__ A, const float* __restrict__ Bp,
                   const float* __restrict__ w, const float* __restrict__ bias,
                   float* __restrict__ Y)
{
  __shared__ float sbuf[32];
  const int row = blockIdx.x;
  const int t   = threadIdx.x;
  const float x = A[(size_t)row*E_ + t] + Bp[(size_t)row*E_ + t];
  // 16-lane DPP reduction; lanes 0 and 16 of each wave hold half-row partials
  float s  = redsum16(x);
  float s2 = redsum16(x*x);
  const int wave = t >> 5, lane = t & 31;
  if ((lane & 15) == 0){
    sbuf[wave*2 + (lane >> 4)]      = s;
    sbuf[16 + wave*2 + (lane >> 4)] = s2;
  }
  __syncthreads();
  float S = 0.f, S2 = 0.f;
#pragma unroll
  for (int i = 0; i < 16; ++i){ S += sbuf[i]; S2 += sbuf[16 + i]; }
  const float mean = S * (1.f/E_);
  const float var  = S2 * (1.f/E_) - mean*mean;
  Y[(size_t)row*E_ + t] = (x - mean) * rsqrtf(var + 1e-5f) * w[t] + bias[t];
}

// qse = sine4(sigmoid(rbs)) * pt   (pt == nullptr -> 1.0)
__global__ void qse_kernel(const float* __restrict__ rbs, const float* __restrict__ pt,
                           float* __restrict__ qse, int n)
{
  int idx = blockIdx.x*256 + threadIdx.x;
  if (idx >= n*E_) return;
  const int tok = idx >> 8;
  const int c   = idx & 255;
  const int blk = c >> 6;     // 0:py 1:px 2:pw 3:ph
  const int j   = c & 63;
  const float x = rbs[tok*4 + 0];
  const float y = rbs[tok*4 + 1];
  const float sx = 1.f/(1.f + __expf(-x));
  const float sy = 1.f/(1.f + __expf(-y));
  const float s  = 6.283185307179586f;
  const float dim_t = __powf(10000.f, (float)(2*(j >> 1)) * (1.f/64.f));
  float val;
  if (blk == 0){
    float a = sy*s/dim_t; val = (j & 1) ? __cosf(a) : __sinf(a);
  } else if (blk == 1){
    float a = sx*s/dim_t; val = (j & 1) ? __cosf(a) : __sinf(a);
  } else {
    float base = (blk == 2) ? sx : sy;
    float a = base*s/dim_t;
    float inner = (j & 1) ? __cosf(a) : __sinf(a);
    val = (j & 1) ? __cosf(inner) : __sinf(inner);
  }
  const float p = pt ? pt[idx] : 1.f;
  qse[idx] = val * p;
}

// ---------------------------------------------------------------------------
// Host orchestration
// ---------------------------------------------------------------------------
static inline void launch_gemm(const float* X, const float* W, const float* b,
                               float* Y, int N, int K, int Mo,
                               float alpha, int relu, hipStream_t s)
{
  dim3 g((N + 127)/128, (Mo + 63)/64);
  if ((K & 31) == 0)
    gemm_bias<1><<<g, 256, 0, s>>>(X, W, b, Y, N, K, Mo, alpha, relu);
  else
    gemm_bias<0><<<g, 256, 0, s>>>(X, W, b, Y, N, K, Mo, alpha, relu);
}

extern "C" void kernel_launch(void* const* d_in, const int* in_sizes, int n_in,
                              void* d_out, int out_size, void* d_ws, size_t ws_size,
                              hipStream_t stream)
{
  (void)in_sizes; (void)n_in; (void)out_size; (void)ws_size;
  auto in = [&](int i){ return (const float*)d_in[i]; };
  const float* tgt       = in(0);
  const float* memory    = in(1);
  const float* pos       = in(2);
  const float* query_pos = in(3);
  const float* box_embed = in(4);
  const float* sa_in_w   = in(5);  const float* sa_in_b  = in(6);
  const float* sa_out_w  = in(7);  const float* sa_out_b = in(8);
  const float* ca_in_w   = in(9);  const float* ca_in_b  = in(10);
  const float* ca_out_w  = in(11); const float* ca_out_b = in(12);
  const float* ca_qpos_w = in(13); const float* ca_qpos_b= in(14);
  const float* ca_kpos_w = in(15); const float* ca_kpos_b= in(16);
  const float* ca_v_w    = in(17); const float* ca_v_b   = in(18);
  const float* ca_kc_w   = in(19); const float* ca_kc_b  = in(20);
  const float* ca_qs_w   = in(21); const float* ca_qs_b  = in(22);
  const float* ca_vp_w   = in(23); const float* ca_vp_b  = in(24);
  const float* lin1_w    = in(25); const float* lin1_b   = in(26);
  const float* lin2_w    = in(27); const float* lin2_b   = in(28);
  const float* lp1_w     = in(29); const float* lp1_b    = in(30);
  const float* lp2_w     = in(31); const float* lp2_b    = in(32);
  const float* norms_w   = in(33); const float* norms_b  = in(34);
  const float* qs_w1 = in(35); const float* qs_b1 = in(36);
  const float* qs_w2 = in(37); const float* qs_b2 = in(38);
  const float* rh_w1 = in(39); const float* rh_b1 = in(40);
  const float* rh_w2 = in(41); const float* rh_b2 = in(42);
  const float* bo_w1 = in(43); const float* bo_b1 = in(44);
  const float* bo_w2 = in(45); const float* bo_b2 = in(46);

  // workspace layout (fp32)
  float* wsf = (float*)d_ws;
  size_t off = 0;
  auto alloc = [&](size_t n){ float* p = wsf + off; off += n; return p; };
  float* out_cur  = alloc((size_t)NTQ*E_);
  float* qbuf     = alloc((size_t)NTQ*E_);   // q / qc
  float* t_buf    = alloc((size_t)NTQ*E_);
  float* tp_buf   = alloc((size_t)NTQ*E_);
  float* t2_buf   = alloc((size_t)NTQ*E_);
  float* t2p_buf  = alloc((size_t)NTQ*E_);
  float* ao_buf   = alloc((size_t)NTQ*E_);   // attention out (pre-proj)
  float* pt_buf   = alloc((size_t)NTQ*E_);
  float* qse_buf  = alloc((size_t)NTQ*E_);   // reused as FFN-out tmp later
  float* qsep_buf = alloc((size_t)NTQ*E_);
  float* qp_buf   = alloc((size_t)NTQ*E_);
  float* kp_buf   = alloc((size_t)NTM*E_);
  float* vp_buf   = alloc((size_t)NTM*E_);
  float* kpos_buf = alloc((size_t)NTM*E_);
  float* vmem_buf = alloc((size_t)NTM*E_);
  float* kk_buf   = alloc((size_t)NTM*E_);
  float* vpos_buf = alloc((size_t)NTM*E_);
  float* hid      = alloc((size_t)NTQ*F_);   // FFN / MLP hidden
  float* rbs_buf  = alloc((size_t)NTQ*4);
  float* rbs_tmp  = alloc((size_t)NTQ*4);

  const float rs = 0.17677669529663687f;     // 1/sqrt(Dh)
  auto add2 = [&](const float* a, const float* b, float* y, int n){
    add2_kernel<<<(n + 255)/256, 256, 0, stream>>>(a, b, y, n);
  };
  auto add_ln = [&](const float* a, const float* b, int l, int ni, float* y){
    add_ln_kernel<<<NTQ, 256, 0, stream>>>(a, b, norms_w + ((size_t)l*6 + ni)*E_,
                                           norms_b + ((size_t)l*6 + ni)*E_, y);
  };
  auto attn = [&](const float* Qp, const float* Kp, const float* Vp, float* O,
                  int Sq, int Sk){
    dim3 g(B_*H_, (Sq + 127)/128);
    attn_kernel<<<g, 256, 0, stream>>>(Qp, Kp, Vp, O, Sq, Sk);
  };

  // rbs = lin(relu(lin(query_pos, rh1)), rh2)
  launch_gemm(query_pos, rh_w1, rh_b1, hid, NTQ, E_, E_, 1.f, 1, stream);
  launch_gemm(hid, rh_w2, rh_b2, rbs_buf, NTQ, E_, 4, 1.f, 0, stream);
  hipMemcpyAsync(out_cur, tgt, (size_t)NTQ*E_*sizeof(float),
                 hipMemcpyDeviceToDevice, stream);

  for (int l = 0; l < L_; ++l){
    // pt (l>0)
    if (l > 0){
      launch_gemm(out_cur, qs_w1, qs_b1, hid, NTQ, E_, E_, 1.f, 1, stream);
      launch_gemm(hid, qs_w2, qs_b2, pt_buf, NTQ, E_, E_, 1.f, 0, stream);
    }
    // rbs += bo-MLP(box_embed)
    launch_gemm(box_embed, bo_w1 + (size_t)l*(E_/2)*4, bo_b1 + (size_t)l*(E_/2),
                hid, NTQ, 4, E_/2, 1.f, 1, stream);
    launch_gemm(hid, bo_w2 + (size_t)l*4*(E_/2), bo_b2 + (size_t)l*4,
                rbs_tmp, NTQ, E_/2, 4, 1.f, 0, stream);
    add2(rbs_buf, rbs_tmp, rbs_buf, NTQ*4);
    // qse = sine4(sigmoid(rbs)) * pt
    qse_kernel<<<(NTQ*E_ + 255)/256, 256, 0, stream>>>(
        rbs_buf, (l == 0) ? nullptr : pt_buf, qse_buf, NTQ);

    // ---- self attention ----
    add2(out_cur, query_pos, qbuf, NTQ*E_);           // q = out + query_pos
    const float* sw = sa_in_w + (size_t)l*3*E_*E_;
    const float* sb = sa_in_b + (size_t)l*3*E_;
    launch_gemm(qbuf,    sw,            sb,         qp_buf, NTQ, E_, E_, rs,  0, stream);
    launch_gemm(qbuf,    sw +   E_*E_,  sb +   E_,  kp_buf, NTQ, E_, E_, 1.f, 0, stream);
    launch_gemm(out_cur, sw + 2*E_*E_,  sb + 2*E_,  vp_buf, NTQ, E_, E_, 1.f, 0, stream);
    attn(qp_buf, kp_buf, vp_buf, ao_buf, NQ_, NQ_);
    launch_gemm(ao_buf, sa_out_w + (size_t)l*E_*E_, sa_out_b + (size_t)l*E_,
                t2_buf, NTQ, E_, E_, 1.f, 0, stream);
    add_ln(out_cur, t2_buf, l, 0, t_buf);             // t = LN(out + sa)

    // ---- cross-attention inputs ----
    const float* qc_ptr;
    const float* kk_ptr;  const float* v_ptr;  const float* kpos_ptr;
    if (l == 0){
      launch_gemm(query_pos, ca_qpos_w, ca_qpos_b, t2_buf, NTQ, E_, E_, 1.f, 0, stream);
      add2(t_buf, t2_buf, qbuf, NTQ*E_);              // qc = t + lin(query_pos)
      qc_ptr = qbuf;
      add2(memory, pos, kk_buf, NTM*E_);              // kk = memory + pos
      kk_ptr = kk_buf;  v_ptr = memory;  kpos_ptr = pos;
    } else {
      qc_ptr = t_buf;
      launch_gemm(pos,    ca_kpos_w + (size_t)l*E_*E_, ca_kpos_b + (size_t)l*E_,
                  kpos_buf, NTM, E_, E_, 1.f, 0, stream);
      launch_gemm(memory, ca_v_w   + (size_t)l*E_*E_, ca_v_b   + (size_t)l*E_,
                  vmem_buf, NTM, E_, E_, 1.f, 0, stream);
      launch_gemm(memory, ca_kc_w  + (size_t)l*E_*E_, ca_kc_b  + (size_t)l*E_,
                  kk_buf,   NTM, E_, E_, 1.f, 0, stream);
      kk_ptr = kk_buf;  v_ptr = vmem_buf;  kpos_ptr = kpos_buf;
    }
    // qse_p
    launch_gemm(qse_buf, ca_qs_w + (size_t)l*E_*E_, ca_qs_b + (size_t)l*E_,
                qsep_buf, NTQ, E_, E_, 1.f, 0, stream);

    // ---- t2 = mha(qc, kk, v) with ca_in weights ----
    const float* cw = ca_in_w + (size_t)l*3*E_*E_;
    const float* cb = ca_in_b + (size_t)l*3*E_;
    launch_gemm(qc_ptr, cw,            cb,        qp_buf, NTQ, E_, E_, rs,  0, stream);
    launch_gemm(kk_ptr, cw +   E_*E_,  cb +  E_,  kp_buf, NTM, E_, E_, 1.f, 0, stream);
    launch_gemm(v_ptr,  cw + 2*E_*E_,  cb + 2*E_, vp_buf, NTM, E_, E_, 1.f, 0, stream);
    attn(qp_buf, kp_buf, vp_buf, ao_buf, NQ_, M_);
    launch_gemm(ao_buf, ca_out_w + (size_t)l*E_*E_, ca_out_b + (size_t)l*E_,
                t2_buf, NTQ, E_, E_, 1.f, 0, stream);

    // ---- t2p = mha(qse_p, kpos, vpos) with same ca_in weights ----
    launch_gemm(memory, ca_vp_w + (size_t)l*E_*E_, ca_vp_b + (size_t)l*E_,
                vpos_buf, NTM, E_, E_, 1.f, 0, stream);
    launch_gemm(qsep_buf, cw,            cb,        qp_buf, NTQ, E_, E_, rs,  0, stream);
    launch_gemm(kpos_ptr, cw +   E_*E_,  cb +  E_,  kp_buf, NTM, E_, E_, 1.f, 0, stream);
    launch_gemm(vpos_buf, cw + 2*E_*E_,  cb + 2*E_, vp_buf, NTM, E_, E_, 1.f, 0, stream);
    attn(qp_buf, kp_buf, vp_buf, ao_buf, NQ_, M_);
    launch_gemm(ao_buf, ca_out_w + (size_t)l*E_*E_, ca_out_b + (size_t)l*E_,
                t2p_buf, NTQ, E_, E_, 1.f, 0, stream);

    // ---- tp path ----
    add_ln(t_buf, t2p_buf, l, 3, tp_buf);             // tp = LN(t + t2p)
    launch_gemm(tp_buf, lp1_w + (size_t)l*F_*E_, lp1_b + (size_t)l*F_,
                hid, NTQ, E_, F_, 1.f, 1, stream);
    launch_gemm(hid, lp2_w + (size_t)l*E_*F_, lp2_b + (size_t)l*E_,
                qse_buf, NTQ, F_, E_, 1.f, 0, stream);  // qse_buf reused as tmp
    add_ln(tp_buf, qse_buf, l, 4, tp_buf);            // tp = LN(tp + FFN)

    // ---- t path ----
    add2(t2_buf, t2p_buf, t2_buf, NTQ*E_);            // t2 += t2p
    add_ln(t_buf, t2_buf, l, 1, t_buf);               // t = LN(t + t2)
    launch_gemm(t_buf, lin1_w + (size_t)l*F_*E_, lin1_b + (size_t)l*F_,
                hid, NTQ, E_, F_, 1.f, 1, stream);
    launch_gemm(hid, lin2_w + (size_t)l*E_*F_, lin2_b + (size_t)l*E_,
                qse_buf, NTQ, F_, E_, 1.f, 0, stream);
    add_ln(t_buf, qse_buf, l, 2, t_buf);              // t = LN(t + FFN)

    add_ln(t_buf, tp_buf, l, 5, out_cur);             // out = LN(t + tp)
  }

  hipMemcpyAsync(d_out, out_cur, (size_t)NTQ*E_*sizeof(float),
                 hipMemcpyDeviceToDevice, stream);
}
